// HeterogeneousGNN_77884936946004
// MI455X (gfx1250) — matrix operand order repaired and checked by reading, not compiled, not run on hardware
//
#include <hip/hip_runtime.h>

typedef unsigned int        u32;
typedef unsigned short      u16;
typedef unsigned long long  u64;
typedef u32   u32x4 __attribute__((ext_vector_type(4)));
typedef u32   u32x2 __attribute__((ext_vector_type(2)));
typedef int   i32x8 __attribute__((ext_vector_type(8)));
typedef int   i32x4 __attribute__((ext_vector_type(4)));
typedef __bf16 v16bf __attribute__((ext_vector_type(16)));
typedef float  v8f   __attribute__((ext_vector_type(8)));

#define KB 32      // batch
#define KN 400     // entities
#define KE 200     // evidences
#define KD 768     // emb dim
#define KEP 224    // E padded to multiple of 32
#define KNP 416    // N padded to multiple of 32
#define TN 48      // 768/16 column tiles

#if __has_builtin(__builtin_amdgcn_tensor_load_to_lds)
#define USE_TDM 1
#else
#define USE_TDM 0
#endif

__device__ __forceinline__ u16 f2bf(float f) {
  u32 u = __float_as_uint(f);
  u += 0x7FFFu + ((u >> 16) & 1u);
  return (u16)(u >> 16);
}
__device__ __forceinline__ float bf2f(u16 h) {
  return __uint_as_float(((u32)h) << 16);
}
__device__ __forceinline__ void wait_tensorcnt0() {
#if __has_builtin(__builtin_amdgcn_s_wait_tensorcnt)
  __builtin_amdgcn_s_wait_tensorcnt(0);
#else
  asm volatile("s_wait_tensorcnt 0x0" ::: "memory");
#endif
}

union FragAB { v16bf v; u32x4 q[2]; };

#if USE_TDM
// ---------------------------------------------------------------------------
// Issue a TDM 2D tile load: 16-dword-wide rows x 64 rows, global (row stride
// strideDw dwords) -> LDS (contiguous).  tensor_dim1 = validRows makes the
// TDM zero-fill rows past the end of the matrix (free M-tail padding).
// Descriptor layout per CDNA5 ISA section 8 (D# groups 0/1; 2D so the
// remaining descriptor groups are zero / disabled).
// This toolchain's builtin takes 6 args (g0, g1, g2, g3, g4, cpol).
// ---------------------------------------------------------------------------
__device__ __forceinline__ void tdm_load_tile(u32 ldsOff, const u16* gtile,
                                              u32 strideDw, u32 validRows) {
  u64 ga = (u64)(uintptr_t)gtile;
  u32x4 g0;
  g0.x = 1u;                                             // count=1, user mode
  g0.y = ldsOff;                                         // lds_addr
  g0.z = (u32)ga;                                        // global_addr[31:0]
  g0.w = (u32)((ga >> 32) & 0x1FFFFFFu) | (2u << 30);    // addr[56:32], type=2
  i32x8 g1;
  g1[0] = (int)(2u << 16);                 // workgroup_mask=0, data_size=4B
  g1[1] = (int)(16u << 16);                // tensor_dim0 = 16 dwords (lo16 in [63:48])
  g1[2] = (int)((validRows & 0xFFFFu) << 16);   // dim0 hi=0, tensor_dim1 lo16
  g1[3] = (int)(((validRows >> 16) & 0xFFFFu) | (16u << 16)); // dim1 hi, tile_dim0=16
  g1[4] = 64;                              // tile_dim1 = 64 rows, tile_dim2=0
  g1[5] = (int)strideDw;                   // tensor_dim0_stride[31:0]
  g1[6] = 0;
  g1[7] = 0;
  i32x4 z4 = {0, 0, 0, 0};
  i32x8 z8 = {0, 0, 0, 0, 0, 0, 0, 0};
  __builtin_amdgcn_tensor_load_to_lds(g0, g1, z4, z4, z8, 0);
}
#endif

// ---------------------------------------------------------------------------
// fp32 -> bf16 convert
// ---------------------------------------------------------------------------
__global__ __launch_bounds__(256) void cvt_kernel(const float* __restrict__ s,
                                                  u16* __restrict__ d, size_t n) {
  size_t i = (size_t)blockIdx.x * 256 + threadIdx.x;
  if (i < n) d[i] = f2bf(s[i]);
}

// ---------------------------------------------------------------------------
// adjacency normalization and padded bf16 transposed layouts (WMMA A-operands)
// ---------------------------------------------------------------------------
__global__ __launch_bounds__(256) void adj_colsum(const float* __restrict__ adj,
                                                  float* __restrict__ cs) {
  int idx = blockIdx.x * 256 + threadIdx.x;
  if (idx >= KB * KN) return;
  int b = idx / KN, n = idx % KN;
  const float* p = adj + (size_t)b * KE * KN + n;
  float s = 0.f;
  for (int e = 0; e < KE; ++e) s += p[(size_t)e * KN];
  cs[idx] = s;
}

__global__ __launch_bounds__(256) void adj_rowsum(const float* __restrict__ adj,
                                                  float* __restrict__ rs) {
  int idx = blockIdx.x * 256 + threadIdx.x;
  if (idx >= KB * KE) return;
  int b = idx / KE, e = idx % KE;
  const float* p = adj + (size_t)b * KE * KN + (size_t)e * KN;
  float s = 0.f;
  for (int n = 0; n < KN; ++n) s += p[n];
  rs[idx] = s;
}

__global__ __launch_bounds__(256) void adj_fillT(const float* __restrict__ adj,
                                                 const float* __restrict__ cs,
                                                 u16* __restrict__ dst) {
  size_t idx = (size_t)blockIdx.x * 256 + threadIdx.x;
  if (idx >= (size_t)KB * KN * KEP) return;
  int ep = (int)(idx % KEP);
  int n  = (int)((idx / KEP) % KN);
  int b  = (int)(idx / ((size_t)KEP * KN));
  float v = 0.f;
  if (ep < KE)
    v = adj[(size_t)b * KE * KN + (size_t)ep * KN + n] / (cs[b * KN + n] + 1e-9f);
  dst[idx] = f2bf(v);
}

__global__ __launch_bounds__(256) void adj_fillNE(const float* __restrict__ adj,
                                                  const float* __restrict__ rs,
                                                  u16* __restrict__ dst) {
  size_t idx = (size_t)blockIdx.x * 256 + threadIdx.x;
  if (idx >= (size_t)KB * KE * KNP) return;
  int np = (int)(idx % KNP);
  int e  = (int)((idx / KNP) % KE);
  int b  = (int)(idx / ((size_t)KNP * KE));
  float v = 0.f;
  if (np < KN)
    v = adj[(size_t)b * KE * KN + (size_t)e * KN + np] / (rs[b * KE + e] + 1e-9f);
  dst[idx] = f2bf(v);
}

// ---------------------------------------------------------------------------
// Pack a row-major [M x 768] matrix (fp32 or bf16 source; rows zero-padded to
// Kpad) into the WMMA B-operand lane layout:
// tile (tk,tn): lane L holds column tn*16+(L&15), K = tk*32+(L>>4)*8 + {0..7, 16..23}
// as 16 contiguous bf16 (two b128 loads in the GEMM).
// ---------------------------------------------------------------------------
__global__ __launch_bounds__(256) void pack_b_kernel(const u16* __restrict__ srcB,
                                                     const float* __restrict__ srcF,
                                                     int M, int Kpad,
                                                     u16* __restrict__ dst,
                                                     size_t srcBatch, size_t dstBatch) {
  int b = blockIdx.y;
  size_t el = (size_t)blockIdx.x * 256 + threadIdx.x;
  size_t H = (size_t)Kpad * KD;
  if (el >= H) return;
  int lh   = (int)(el & 15);
  int lane = (int)((el >> 4) & 31);
  int tile = (int)(el >> 9);
  int tn = tile % TN, tk = tile / TN;
  int col = tn * 16 + (lane & 15);
  int k   = tk * 32 + (lane >> 4) * 8 + (lh >> 3) * 16 + (lh & 7);
  u16 v = 0;
  if (k < M) {
    size_t si = (size_t)b * srcBatch + (size_t)k * KD + col;
    v = srcF ? f2bf(srcF[si]) : srcB[si];
  }
  dst[(size_t)b * dstBatch + el] = v;
}

// ---------------------------------------------------------------------------
// Generic batched WMMA GEMM:  C[b] (M x 768) = A[b](M x 32*Kt bf16) @ Bp
//   epilogue: (+bias[col]) (+addmat[b,row,col]) (*rowscale[b,row]) (relu)
//   outputs:  fp32 row-major and/or bf16 row-major.
// Block = 256 threads = 8 wave32 waves.  Block tile 64x128; wave tile 32x32
// (2 A-frags x 2 B-frags -> 4 WMMAs per K-step).  A panels (64 rows x 32 K)
// are DMA'd into double-buffered LDS by the Tensor Data Mover.
// ---------------------------------------------------------------------------
__global__ __launch_bounds__(256) void wmma_gemm(
    const u16* __restrict__ A, int strideA, int M, int Kt,
    const u16* __restrict__ Bp, size_t bBatch,
    const float* __restrict__ bias, const float* __restrict__ rowscale,
    const float* __restrict__ addmat,
    float* __restrict__ outF, u16* __restrict__ outB, int relu) {
  __shared__ u16 sA[2][64 * 32];

  const int tid  = threadIdx.x;
  const int lane = tid & 31;
  const int w    = tid >> 5;
  const int wr   = w & 1;        // 0..1 : 32-row group within block
  const int wc   = w >> 1;       // 0..3 : 32-col group within block
  const int b    = blockIdx.z;
  const int m0   = blockIdx.x * 64;
  const int tn0  = blockIdx.y * 8 + wc * 2;     // first 16-col tile of wave
  const u16* Ab  = A + (size_t)b * (size_t)M * strideA;
  const u16* Bb  = Bp + (size_t)b * bBatch;
  const int half = lane >> 4;
  const int l15  = lane & 15;

  v8f c00 = {}, c01 = {}, c10 = {}, c11 = {};

#if USE_TDM
  const u32 strideDw = (u32)strideA >> 1;
  const u32 validRows = (u32)(M - m0);
  const u32 ldsBase = (u32)(uintptr_t)(&sA[0][0]);
  if (w == 0)
    tdm_load_tile(ldsBase, Ab + (size_t)m0 * strideA, strideDw, validRows);
#else
  const int lr = tid >> 2;             // 0..63 panel row for coop load
  const int lc = (tid & 3) * 8;        // half offset (16B chunks)
#endif

  for (int kt = 0; kt < Kt; ++kt) {
    const int cur = kt & 1;
#if USE_TDM
    wait_tensorcnt0();
    __syncthreads();
    if (w == 0 && kt + 1 < Kt)
      tdm_load_tile(ldsBase + (u32)((1 ^ cur) * 64 * 32 * 2),
                    Ab + (size_t)m0 * strideA + (size_t)(kt + 1) * 32,
                    strideDw, validRows);
#else
    {
      u32x4 d = (u32x4)(0u);
      int grow = m0 + lr;
      if (grow < M)
        d = *(const u32x4*)(Ab + (size_t)grow * strideA + kt * 32 + lc);
      *(u32x4*)(&sA[cur][lr * 32 + lc]) = d;
      __syncthreads();
    }
#endif
    FragAB a0, a1, b0, b1;
    const u16* pA0 = &sA[cur][(wr * 32 + l15) * 32 + half * 8];
    const u16* pA1 = &sA[cur][(wr * 32 + 16 + l15) * 32 + half * 8];
    a0.q[0] = *(const u32x4*)(pA0);
    a0.q[1] = *(const u32x4*)(pA0 + 16);
    a1.q[0] = *(const u32x4*)(pA1);
    a1.q[1] = *(const u32x4*)(pA1 + 16);
    const u32x4* bp0 = (const u32x4*)(Bb + (((size_t)kt * TN + tn0) * 32 + lane) * 16);
    const u32x4* bp1 = (const u32x4*)(Bb + (((size_t)kt * TN + tn0 + 1) * 32 + lane) * 16);
    b0.q[0] = bp0[0]; b0.q[1] = bp0[1];
    b1.q[0] = bp1[0]; b1.q[1] = bp1[1];

    c00 = __builtin_amdgcn_wmma_f32_16x16x32_bf16(false, a0.v, false, b0.v, (short)0, c00, false, false);
    c01 = __builtin_amdgcn_wmma_f32_16x16x32_bf16(false, a0.v, false, b1.v, (short)0, c01, false, false);
    c10 = __builtin_amdgcn_wmma_f32_16x16x32_bf16(false, a1.v, false, b0.v, (short)0, c10, false, false);
    c11 = __builtin_amdgcn_wmma_f32_16x16x32_bf16(false, a1.v, false, b1.v, (short)0, c11, false, false);
#if !USE_TDM
    __syncthreads();
#endif
  }

  // epilogue: C VGPR r -> row = base + r + (lane>=16 ? 8 : 0), col = lane&15
  const size_t zoff = (size_t)b * (size_t)M * KD;
  float cv[2][2][8];
  for (int r = 0; r < 8; ++r) {
    cv[0][0][r] = c00[r]; cv[0][1][r] = c01[r];
    cv[1][0][r] = c10[r]; cv[1][1][r] = c11[r];
  }
  for (int i = 0; i < 2; ++i) {
    for (int j = 0; j < 2; ++j) {
      const int colg = (tn0 + j) * 16 + l15;
      for (int r = 0; r < 8; ++r) {
        int row = m0 + wr * 32 + i * 16 + (half << 3) + r;
        if (row < M) {
          float v = cv[i][j][r];
          if (bias)     v += bias[colg];
          if (addmat)   v += addmat[zoff + (size_t)row * KD + colg];
          if (rowscale) v *= rowscale[b * M + row];
          if (relu)     v = v > 0.f ? v : 0.f;
          size_t o = zoff + (size_t)row * KD + colg;
          if (outF) outF[o] = v;
          if (outB) outB[o] = f2bf(v);
        }
      }
    }
  }
}

// ---------------------------------------------------------------------------
// u[b][d] = sum_e W[d][e] * sr[b][e]   (+ optional cbuf[b] = bvec . sr[b])
// ---------------------------------------------------------------------------
__global__ __launch_bounds__(256) void gemv_u(const float* __restrict__ W,
                                              const float* __restrict__ bvec,
                                              const float* __restrict__ sr,
                                              float* __restrict__ u,
                                              float* __restrict__ cbuf) {
  int idx = blockIdx.x * 256 + threadIdx.x;
  if (idx >= KB * KD) return;
  int b = idx / KD, d = idx % KD;
  const float* srb = sr + (size_t)b * KD;
  const float* wr  = W + (size_t)d * KD;
  float s = 0.f;
  for (int e = 0; e < KD; ++e) s += wr[e] * srb[e];
  u[idx] = s;
  if (cbuf && d == 0) {
    float cc = 0.f;
    if (bvec)
      for (int e = 0; e < KD; ++e) cc += bvec[e] * srb[e];
    cbuf[b] = cc;
  }
}

// ---------------------------------------------------------------------------
// s[b][m] = X[b][m] . u[b] + cbuf[b] + cscalar;  optional softmax over m.
// ---------------------------------------------------------------------------
__global__ __launch_bounds__(256) void scores_kernel(
    const u16* __restrict__ Xbf, int M, const float* __restrict__ u,
    const float* __restrict__ cbuf, const float* __restrict__ cscalar,
    float* __restrict__ out, int outStride, int doSoftmax) {
  __shared__ float red[256];
  __shared__ float tmp[KNP];
  int b = blockIdx.x, tid = threadIdx.x;
  const u16* X = Xbf + (size_t)b * M * KD;
  const float* ub = u + (size_t)b * KD;
  float cadd = 0.f;
  if (cbuf) cadd += cbuf[b];
  if (cscalar) cadd += cscalar[0];
  for (int m = tid; m < M; m += 256) {
    float s = cadd;
    const u16* xr = X + (size_t)m * KD;
    for (int d = 0; d < KD; ++d) s += bf2f(xr[d]) * ub[d];
    tmp[m] = s;
  }
  __syncthreads();
  if (!doSoftmax) {
    for (int m = tid; m < M; m += 256) out[(size_t)b * outStride + m] = tmp[m];
    return;
  }
  float mx = -3.4e38f;
  for (int m = tid; m < M; m += 256) mx = fmaxf(mx, tmp[m]);
  red[tid] = mx;
  __syncthreads();
  for (int s = 128; s > 0; s >>= 1) {
    if (tid < s) red[tid] = fmaxf(red[tid], red[tid + s]);
    __syncthreads();
  }
  mx = red[0];
  __syncthreads();
  float sm = 0.f;
  for (int m = tid; m < M; m += 256) {
    float e = __expf(tmp[m] - mx);
    tmp[m] = e;
    sm += e;
  }
  red[tid] = sm;
  __syncthreads();
  for (int s = 128; s > 0; s >>= 1) {
    if (tid < s) red[tid] += red[tid + s];
    __syncthreads();
  }
  float inv = 1.0f / red[0];
  for (int m = tid; m < M; m += 256) out[(size_t)b * outStride + m] = tmp[m] * inv;
}

// ---------------------------------------------------------------------------
extern "C" void kernel_launch(void* const* d_in, const int* in_sizes, int n_in,
                              void* d_out, int out_size, void* d_ws, size_t ws_size,
                              hipStream_t stream) {
  (void)in_sizes; (void)n_in; (void)out_size; (void)ws_size;
  const float* ent_emb = (const float*)d_in[0];
  const float* ev_emb  = (const float*)d_in[1];
  const float* sr      = (const float*)d_in[2];
  const float* adj     = (const float*)d_in[3];
  const float* W_att[2] = { (const float*)d_in[4], (const float*)d_in[6] };   // ev_att, ent_att
  const float* b_att[2] = { (const float*)d_in[5], (const float*)d_in[7] };
  const float* W_gemm[4] = { (const float*)d_in[8],  (const float*)d_in[10],
                             (const float*)d_in[12], (const float*)d_in[14] }; // ev_ent, ent_ent, ent_ev, ev_ev
  const float* b_gemm[4] = { (const float*)d_in[9],  (const float*)d_in[11],
                             (const float*)d_in[13], (const float*)d_in[15] };
  const float* W_ans = (const float*)d_in[16];
  const float* b_ans = (const float*)d_in[17];
  const float* W_evb = (const float*)d_in[18];
  const float* b_evb = (const float*)d_in[19];
  float* out = (float*)d_out;

  char* ws = (char*)d_ws;
  size_t off = 0;
  auto alloc = [&](size_t bytes) -> char* {
    char* p = ws + off;
    off += (bytes + 255) & ~(size_t)255;
    return p;
  };

  u16* entA0 = (u16*)alloc((size_t)KB * KN * KD * 2);
  u16* entA1 = (u16*)alloc((size_t)KB * KN * KD * 2);
  u16* evA0  = (u16*)alloc((size_t)KB * KE * KD * 2);
  u16* evA1  = (u16*)alloc((size_t)KB * KE * KD * 2);
  u16* Wp    = (u16*)alloc((size_t)12 * KD * KD * 2);
  u16* adjT  = (u16*)alloc((size_t)KB * KN * KEP * 2);
  u16* adjNE = (u16*)alloc((size_t)KB * KE * KNP * 2);
  float* cs  = (float*)alloc((size_t)KB * KN * 4);
  float* rs  = (float*)alloc((size_t)KB * KE * 4);
  u16* msg_rm   = (u16*)alloc((size_t)KB * KN * KD * 2);
  u16* evmsg_p  = (u16*)alloc((size_t)KB * KEP * KD * 2);
  u16* entmsg_p = (u16*)alloc((size_t)KB * KNP * KD * 2);
  float* Gent = (float*)alloc((size_t)KB * KN * KD * 4);
  float* Gev  = (float*)alloc((size_t)KB * KE * KD * 4);
  float* u_buf = (float*)alloc((size_t)KB * KD * 4);
  float* c_buf = (float*)alloc((size_t)KB * 4);
  float* att_ev  = (float*)alloc((size_t)KB * KE * 4);
  float* att_ent = (float*)alloc((size_t)KB * KN * 4);

  // bf16 copies of the initial states
  cvt_kernel<<<(KB * KN * KD + 255) / 256, 256, 0, stream>>>(ent_emb, entA0, (size_t)KB * KN * KD);
  cvt_kernel<<<(KB * KE * KD + 255) / 256, 256, 0, stream>>>(ev_emb, evA0, (size_t)KB * KE * KD);

  // normalized adjacency operands
  adj_colsum<<<(KB * KN + 255) / 256, 256, 0, stream>>>(adj, cs);
  adj_rowsum<<<(KB * KE + 255) / 256, 256, 0, stream>>>(adj, rs);
  adj_fillT<<<(KB * KN * KEP + 255) / 256, 256, 0, stream>>>(adj, cs, adjT);
  adj_fillNE<<<(KB * KE * KNP + 255) / 256, 256, 0, stream>>>(adj, rs, adjNE);

  // pack the 12 GEMM weights into WMMA B-operand layout
  for (int f = 0; f < 4; ++f)
    for (int i = 0; i < 3; ++i)
      pack_b_kernel<<<dim3((KD * KD + 255) / 256, 1), 256, 0, stream>>>(
          nullptr, W_gemm[f] + (size_t)i * KD * KD, KD, KD,
          Wp + (size_t)(f * 3 + i) * KD * KD, 0, 0);

  u16 *entCur = entA0, *entNxt = entA1, *evCur = evA0, *evNxt = evA1;
  const int Ktd = KD / 32;
  const dim3 gEnt(7, 6, KB);   // ceil(400/64), 768/128, B
  const dim3 gEv(4, 6, KB);    // ceil(200/64)

  for (int i = 0; i < 3; ++i) {
    // attention over evidences / entities: att = softmax(x.(W@sr) + b.sr)
    gemv_u<<<(KB * KD + 255) / 256, 256, 0, stream>>>(
        W_att[0] + (size_t)i * KD * KD, b_att[0] + (size_t)i * KD, sr, u_buf, c_buf);
    scores_kernel<<<KB, 256, 0, stream>>>(evCur, KE, u_buf, c_buf, nullptr, att_ev, KE, 1);
    gemv_u<<<(KB * KD + 255) / 256, 256, 0, stream>>>(
        W_att[1] + (size_t)i * KD * KD, b_att[1] + (size_t)i * KD, sr, u_buf, c_buf);
    scores_kernel<<<KB, 256, 0, stream>>>(entCur, KN, u_buf, c_buf, nullptr, att_ent, KN, 1);

    // ev_msg = (ev @ W_ev_ent + b) * att_ev  -> bf16 row-major, then pack
    wmma_gemm<<<gEv, 256, 0, stream>>>(
        evCur, KD, KE, Ktd, Wp + (size_t)(0 * 3 + i) * KD * KD, 0,
        b_gemm[0] + (size_t)i * KD, att_ev, nullptr, nullptr, msg_rm, 0);
    pack_b_kernel<<<dim3((KEP * KD + 255) / 256, KB), 256, 0, stream>>>(
        msg_rm, nullptr, KE, KEP, evmsg_p, (size_t)KE * KD, (size_t)KEP * KD);

    // ent_msg = (ent @ W_ent_ev + b) * att_ent -> bf16 row-major, then pack
    wmma_gemm<<<gEnt, 256, 0, stream>>>(
        entCur, KD, KN, Ktd, Wp + (size_t)(2 * 3 + i) * KD * KD, 0,
        b_gemm[2] + (size_t)i * KD, att_ent, nullptr, nullptr, msg_rm, 0);
    pack_b_kernel<<<dim3((KNP * KD + 255) / 256, KB), 256, 0, stream>>>(
        msg_rm, nullptr, KN, KNP, entmsg_p, (size_t)KN * KD, (size_t)KNP * KD);

    // self transforms
    wmma_gemm<<<gEnt, 256, 0, stream>>>(
        entCur, KD, KN, Ktd, Wp + (size_t)(1 * 3 + i) * KD * KD, 0,
        b_gemm[1] + (size_t)i * KD, nullptr, nullptr, Gent, nullptr, 0);
    wmma_gemm<<<gEv, 256, 0, stream>>>(
        evCur, KD, KE, Ktd, Wp + (size_t)(3 * 3 + i) * KD * KD, 0,
        b_gemm[3] + (size_t)i * KD, nullptr, nullptr, Gev, nullptr, 0);

    // ent_new = relu(adjT_en @ ev_msg + Gent)
    wmma_gemm<<<gEnt, 256, 0, stream>>>(
        adjT, KEP, KN, KEP / 32, evmsg_p, (size_t)KEP * KD,
        nullptr, nullptr, Gent, nullptr, entNxt, 1);
    // ev_new = relu(adj_ne @ ent_msg + Gev)
    wmma_gemm<<<gEv, 256, 0, stream>>>(
        adjNE, KNP, KE, KNP / 32, entmsg_p, (size_t)KNP * KD,
        nullptr, nullptr, Gev, nullptr, evNxt, 1);

    u16* t;
    t = entCur; entCur = entNxt; entNxt = t;
    t = evCur;  evCur  = evNxt;  evNxt  = t;
  }

  // bilinear heads: ent.(W_ans@sr)+b_ans  and  ev.(W_evb@sr)+b_evb
  gemv_u<<<(KB * KD + 255) / 256, 256, 0, stream>>>(W_ans, nullptr, sr, u_buf, nullptr);
  scores_kernel<<<KB, 256, 0, stream>>>(entCur, KN, u_buf, nullptr, b_ans, out, KN + KE, 0);
  gemv_u<<<(KB * KD + 255) / 256, 256, 0, stream>>>(W_evb, nullptr, sr, u_buf, nullptr);
  scores_kernel<<<KB, 256, 0, stream>>>(evCur, KE, u_buf, nullptr, b_evb, out + KN, KN + KE, 0);
}
